// GPT_24455543783536
// MI455X (gfx1250) — compile-verified, matrix operand order checked by reading, not run on hardware
//
#include <hip/hip_runtime.h>
#include <math.h>

typedef __attribute__((ext_vector_type(16))) __bf16 v16bf;
typedef __attribute__((ext_vector_type(8)))  float  v8f;

#define Bc 2
#define Sc 1024
#define Ec 1024
#define Hc 16
#define Dc 64
#define Fc 4096
#define Vc 50257
#define Lc 4

// round-to-nearest-even f32 -> bf16
__device__ __forceinline__ unsigned short f2bf(float f) {
  unsigned u = __float_as_uint(f);
  u += 0x7fffu + ((u >> 16) & 1u);
  return (unsigned short)(u >> 16);
}
__device__ __forceinline__ unsigned int pk2bf(float a, float b) {
  return (unsigned int)f2bf(a) | ((unsigned int)f2bf(b) << 16);
}

// Async 16-byte copy global -> LDS (CDNA5 GLOBAL_LOAD_ASYNC_TO_LDS_B128, ASYNCcnt).
// LDS operand: low 32 bits of the generic shared-pointer == LDS byte offset.
__device__ __forceinline__ void async_b128(void* lds_ptr, const void* gptr) {
  asm volatile("global_load_async_to_lds_b128 %0, %1, off"
               :: "v"((unsigned)(unsigned long long)lds_ptr),
                  "v"((unsigned long long)gptr)
               : "memory");
}
__device__ __forceinline__ void wait_async() {
  asm volatile("s_wait_asynccnt 0x0" ::: "memory");
}

// Load one 16x32 bf16 WMMA A/B fragment row from a contiguous 32-bf16 chunk.
// lanes 0-15: K=0..7 (VGPR0-3), K=16..23 (VGPR4-7); lanes 16-31: K=8..15, K=24..31.
__device__ __forceinline__ v16bf load_frag(const unsigned short* row, int hi) {
  const unsigned int* r32 = (const unsigned int*)row;
  union { v16bf v; unsigned int u[8]; } f;
  int b = hi ? 4 : 0;
#pragma unroll
  for (int j = 0; j < 4; ++j) { f.u[j] = r32[b + j]; f.u[4 + j] = r32[8 + b + j]; }
  return f.v;
}

// ------------- one-time weight conversion: W[K,N] f32 -> Wb[N,K] bf16 (transpose) -------------
__global__ __launch_bounds__(256) void conv_t_kernel(const float* __restrict__ W,
                                                     unsigned short* __restrict__ Wb,
                                                     int K, int N) {
  __shared__ unsigned short t[64 * 66];
  int tid = threadIdx.x;
  int n0 = blockIdx.x * 64, k0 = blockIdx.y * 64;
  int c = tid & 63, r0 = tid >> 6;            // read: c along n, 4 k-rows per pass
#pragma unroll
  for (int rr = 0; rr < 64; rr += 4) {
    int k = k0 + rr + r0;
    t[c * 66 + rr + r0] = f2bf(W[(size_t)k * N + n0 + c]);
  }
  __syncthreads();
  int kk = (tid & 31) * 2, nr = tid >> 5;     // write: 2 k per thread, 8 n-rows per pass
#pragma unroll
  for (int nn = 0; nn < 64; nn += 8) {
    int n = nn + nr;
    unsigned int p = *(const unsigned int*)&t[n * 66 + kk];
    *(unsigned int*)&Wb[(size_t)(n0 + n) * K + k0 + kk] = p;
  }
}

// ------------- one-time plain conversion (head_W is already [N,K]) -------------
__global__ __launch_bounds__(256) void conv_kernel(const float* __restrict__ W,
                                                   unsigned short* __restrict__ Wb,
                                                   long long n4) {
  long long i = ((long long)blockIdx.x * 256 + threadIdx.x);
  if (i >= n4) return;
  const float4 v = *(const float4*)&W[i * 4];
  uint2 p; p.x = pk2bf(v.x, v.y); p.y = pk2bf(v.z, v.w);
  *(uint2*)&Wb[i * 4] = p;
}

// ---------------- embedding: x = tok_emb[idx] + pos_emb (fp32 residual stream) ----------------
__global__ void embed_kernel(const int* __restrict__ idx, const float* __restrict__ tok,
                             const float* __restrict__ pos, float* __restrict__ x) {
  int row = blockIdx.x;
  int s = row & (Sc - 1);
  int t = idx[row];
  int c = threadIdx.x * 4;
  float4 a = *(const float4*)&tok[(size_t)t * Ec + c];
  float4 p = *(const float4*)&pos[(size_t)s * Ec + c];
  float4 o; o.x = a.x + p.x; o.y = a.y + p.y; o.z = a.z + p.z; o.w = a.w + p.w;
  *(float4*)&x[(size_t)row * Ec + c] = o;
}

// ---------------- layernorm over E=1024: fp32 in -> bf16 out ----------------
__global__ void ln_kernel(const float* __restrict__ x, const float* __restrict__ w,
                          const float* __restrict__ b, unsigned short* __restrict__ out) {
  int row = blockIdx.x;
  int tid = threadIdx.x;
  int c = tid * 4;
  const float4 v = *(const float4*)&x[(size_t)row * Ec + c];
  __shared__ float red[256];
  red[tid] = v.x + v.y + v.z + v.w; __syncthreads();
  for (int o = 128; o > 0; o >>= 1) { if (tid < o) red[tid] += red[tid + o]; __syncthreads(); }
  float mean = red[0] * (1.f / Ec);
  __syncthreads();
  float dx = v.x - mean, dy = v.y - mean, dz = v.z - mean, dw = v.w - mean;
  red[tid] = dx * dx + dy * dy + dz * dz + dw * dw; __syncthreads();
  for (int o = 128; o > 0; o >>= 1) { if (tid < o) red[tid] += red[tid + o]; __syncthreads(); }
  float inv = rsqrtf(red[0] * (1.f / Ec) + 1e-5f);
  const float4 wv = *(const float4*)&w[c];
  const float4 bv = *(const float4*)&b[c];
  uint2 p;
  p.x = pk2bf(dx * inv * wv.x + bv.x, dy * inv * wv.y + bv.y);
  p.y = pk2bf(dz * inv * wv.z + bv.z, dw * inv * wv.w + bv.w);
  *(uint2*)&out[(size_t)row * Ec + c] = p;
}

// ---------------- GEMM: C = act(A[M,K]bf16 @ Wb[N,K]bf16^T + bias) [+ residual] ----------------
// 128x128 tile, 8 waves (4x2), 2x4 WMMA accs per wave.
// Staging: async global->LDS b128 copies (one per thread per side), LDS double-buffered;
// copies for tile k+1 overlap WMMAs on tile k, drained with s_wait_asynccnt before the barrier.
__global__ __launch_bounds__(256) void gemm_wmma(
    const unsigned short* __restrict__ A, const unsigned short* __restrict__ Wb,
    const float* __restrict__ bias, const float* __restrict__ residual,
    void* __restrict__ Cp, int M, int N, int K, int doGelu, int outBf16) {
  __shared__ unsigned short As[2][128 * 32];   // [m][k] bf16
  __shared__ unsigned short Ws[2][128 * 32];   // [n][k] bf16
  int tid = threadIdx.x;
  int lane = tid & 31, waveId = tid >> 5;
  int wm = waveId >> 1, wn = waveId & 1;
  int m0 = blockIdx.y * 128, n0 = blockIdx.x * 128;
  int hi = lane >> 4, lr = lane & 15;

  int rT = tid >> 1, cT = (tid & 1) * 16;      // one b128 chunk per thread per side
  int nB = n0 + rT; if (nB >= N) nB = N - 1;   // clamp: junk cols are never stored

  auto stage = [&](int k0, int buf) {
    async_b128(&As[buf][rT * 32 + cT], &A[(size_t)(m0 + rT) * K + k0 + cT]);
    async_b128(&Ws[buf][rT * 32 + cT], &Wb[(size_t)nB * K + k0 + cT]);
  };

  v8f acc[2][4] = {};
  auto compute = [&](int buf) {
    v16bf af[2], bf[4];
#pragma unroll
    for (int t = 0; t < 2; ++t)
      af[t] = load_frag(&As[buf][(wm * 32 + t * 16 + lr) * 32], hi);
#pragma unroll
    for (int u = 0; u < 4; ++u)
      bf[u] = load_frag(&Ws[buf][(wn * 64 + u * 16 + lr) * 32], hi);
#pragma unroll
    for (int t = 0; t < 2; ++t)
#pragma unroll
      for (int u = 0; u < 4; ++u)
        acc[t][u] = __builtin_amdgcn_wmma_f32_16x16x32_bf16(
            false, af[t], false, bf[u], (short)0, acc[t][u], false, false);
  };

  int nk = K / 32;
  stage(0, 0);
  wait_async();
  __syncthreads();
  for (int ks = 0; ks < nk; ++ks) {
    int nxt = ks + 1;
    if (nxt < nk) stage(nxt * 32, nxt & 1);    // async copy overlaps compute below
    compute(ks & 1);
    wait_async();
    __syncthreads();
  }

  float* Cf = (float*)Cp;
  unsigned short* Cb = (unsigned short*)Cp;
#pragma unroll
  for (int t = 0; t < 2; ++t) {
#pragma unroll
    for (int u = 0; u < 4; ++u) {
      int n = n0 + wn * 64 + u * 16 + lr;
      if (n >= N) continue;
      int mb = m0 + wm * 32 + t * 16 + hi * 8;
      float bvv = bias ? bias[n] : 0.f;
      union { v8f v; float f[8]; } a; a.v = acc[t][u];
#pragma unroll
      for (int r = 0; r < 8; ++r) {
        float val = a.f[r] + bvv;
        if (doGelu) val = 0.5f * val * (1.f + erff(val * 0.70710678118654752f));
        size_t off = (size_t)(mb + r) * N + n;
        if (outBf16) {
          Cb[off] = f2bf(val);
        } else {
          if (residual) val += residual[off];
          Cf[off] = val;
        }
      }
    }
  }
}

// ---------------- attention scores: scale * Q·K^T per (b,h), lower-tri blocks only ----------------
__global__ __launch_bounds__(256) void attn_scores(
    const unsigned short* __restrict__ q, const unsigned short* __restrict__ k,
    float* __restrict__ scores, float scale) {
  if (blockIdx.x > blockIdx.y) return;  // fully-masked tile (uniform exit)
  __shared__ unsigned short Qs[128 * 64];
  __shared__ unsigned short Ks[128 * 64];
  int tid = threadIdx.x;
  int bh = blockIdx.z, b = bh / Hc, h = bh % Hc;
  int s0 = blockIdx.y * 128, t0 = blockIdx.x * 128;
  int rS = tid >> 2, cS = (tid & 3) * 16;      // 64 rows/pass, b128 chunks
#pragma unroll
  for (int rr = 0; rr < 128; rr += 64) {
    async_b128(&Qs[(rr + rS) * 64 + cS],
               &q[((size_t)(b * Sc + s0 + rr + rS)) * Ec + h * Dc + cS]);
    async_b128(&Ks[(rr + rS) * 64 + cS],
               &k[((size_t)(b * Sc + t0 + rr + rS)) * Ec + h * Dc + cS]);
  }
  wait_async();
  __syncthreads();
  int lane = tid & 31, waveId = tid >> 5;
  int wm = waveId >> 1, wn = waveId & 1;
  int hi = lane >> 4, lr = lane & 15;
  v8f acc[2][4] = {};
#pragma unroll
  for (int kk = 0; kk < 2; ++kk) {
    v16bf af[2], bf[4];
#pragma unroll
    for (int t = 0; t < 2; ++t)
      af[t] = load_frag(&Qs[(wm * 32 + t * 16 + lr) * 64 + kk * 32], hi);
#pragma unroll
    for (int u = 0; u < 4; ++u)
      bf[u] = load_frag(&Ks[(wn * 64 + u * 16 + lr) * 64 + kk * 32], hi);
#pragma unroll
    for (int t = 0; t < 2; ++t)
#pragma unroll
      for (int u = 0; u < 4; ++u)
        acc[t][u] = __builtin_amdgcn_wmma_f32_16x16x32_bf16(
            false, af[t], false, bf[u], (short)0, acc[t][u], false, false);
  }
  size_t base = (size_t)bh * Sc * Sc;
#pragma unroll
  for (int t = 0; t < 2; ++t) {
#pragma unroll
    for (int u = 0; u < 4; ++u) {
      union { v8f v; float f[8]; } a; a.v = acc[t][u];
      int tc = t0 + wn * 64 + u * 16 + lr;
      int sb = s0 + wm * 32 + t * 16 + hi * 8;
#pragma unroll
      for (int r = 0; r < 8; ++r)
        scores[base + (size_t)(sb + r) * Sc + tc] = scale * a.f[r];
    }
  }
}

// ---------------- causal softmax: fp32 scores row -> bf16 probs row (zeros for t>s) ----------------
__global__ void softmax_kernel(const float* __restrict__ sc, unsigned short* __restrict__ pr) {
  int s = blockIdx.x;
  size_t base = (size_t)blockIdx.y * Sc * Sc + (size_t)s * Sc;
  int tid = threadIdx.x;
  int t0 = tid * 4;
  float4 v = *(const float4*)&sc[base + t0];
  float vv[4] = {v.x, v.y, v.z, v.w};
  float mx = -3.4e38f;
#pragma unroll
  for (int j = 0; j < 4; ++j) {
    if (t0 + j > s) vv[j] = -3.4e38f;
    mx = fmaxf(mx, vv[j]);
  }
  __shared__ float red[256];
  red[tid] = mx; __syncthreads();
  for (int o = 128; o > 0; o >>= 1) { if (tid < o) red[tid] = fmaxf(red[tid], red[tid + o]); __syncthreads(); }
  float rowmax = red[0];
  __syncthreads();
  float sum = 0.f;
#pragma unroll
  for (int j = 0; j < 4; ++j) {
    vv[j] = (t0 + j <= s) ? __expf(vv[j] - rowmax) : 0.f;
    sum += vv[j];
  }
  red[tid] = sum; __syncthreads();
  for (int o = 128; o > 0; o >>= 1) { if (tid < o) red[tid] += red[tid + o]; __syncthreads(); }
  float inv = 1.f / red[0];
  uint2 p;
  p.x = pk2bf(vv[0] * inv, vv[1] * inv);
  p.y = pk2bf(vv[2] * inv, vv[3] * inv);
  *(uint2*)&pr[base + t0] = p;
}

// ---------------- AV: y = probs @ V (all bf16); K-loop truncated at diagonal; double-buffered ----------------
__global__ __launch_bounds__(256) void attn_av(
    const unsigned short* __restrict__ probs, const unsigned short* __restrict__ v,
    unsigned short* __restrict__ y) {
  __shared__ unsigned short As[2][128 * 32];   // [m][t] bf16
  __shared__ unsigned short Vs[2][64 * 32];    // [d][t] bf16
  int tid = threadIdx.x;
  int bh = blockIdx.y, b = bh / Hc, h = bh % Hc;
  int m0 = blockIdx.x * 128;
  const unsigned short* ab = probs + (size_t)bh * Sc * Sc;
  int lane = tid & 31, waveId = tid >> 5;
  int hi = lane >> 4, lr = lane & 15;

  int rA = tid >> 1, cA = (tid & 1) * 16;      // probs: one b128 chunk per thread (async)
  int cV = tid >> 4, nV = (tid & 15) * 4;      // V: 4 bf16 along d, 16 t-cols/pass (transpose)

  auto stage = [&](int tk, int buf) {
    async_b128(&As[buf][rA * 32 + cA], &ab[(size_t)(m0 + rA) * Sc + tk + cA]);
#pragma unroll
    for (int cc = 0; cc < 32; cc += 16) {
      int c = cV + cc;
      ushort4 vv = *(const ushort4*)&v[((size_t)(b * Sc + tk + c)) * Ec + h * Dc + nV];
      Vs[buf][(nV + 0) * 32 + c] = vv.x;
      Vs[buf][(nV + 1) * 32 + c] = vv.y;
      Vs[buf][(nV + 2) * 32 + c] = vv.z;
      Vs[buf][(nV + 3) * 32 + c] = vv.w;
    }
  };

  v8f acc[4] = {};
  int ksteps = (m0 >> 5) + 4;                  // probs rows are zero past the diagonal
  stage(0, 0);
  wait_async();
  __syncthreads();
  for (int ks = 0; ks < ksteps; ++ks) {
    int nxt = ks + 1;
    if (nxt < ksteps) stage(nxt * 32, nxt & 1);
    v16bf af = load_frag(&As[ks & 1][(waveId * 16 + lr) * 32], hi);
#pragma unroll
    for (int u = 0; u < 4; ++u) {
      v16bf bf = load_frag(&Vs[ks & 1][(u * 16 + lr) * 32], hi);
      acc[u] = __builtin_amdgcn_wmma_f32_16x16x32_bf16(
          false, af, false, bf, (short)0, acc[u], false, false);
    }
    wait_async();
    __syncthreads();
  }
#pragma unroll
  for (int u = 0; u < 4; ++u) {
    union { v8f v8; float f[8]; } a; a.v8 = acc[u];
    int d = u * 16 + lr;
#pragma unroll
    for (int r = 0; r < 8; ++r) {
      int m = m0 + waveId * 16 + hi * 8 + r;
      y[((size_t)(b * Sc + m)) * Ec + h * Dc + d] = f2bf(a.f[r]);
    }
  }
}

extern "C" void kernel_launch(void* const* d_in, const int* in_sizes, int n_in,
                              void* d_out, int out_size, void* d_ws, size_t ws_size,
                              hipStream_t stream) {
  const int*   idx  = (const int*)d_in[0];
  const float* tok  = (const float*)d_in[1];
  const float* pos  = (const float*)d_in[2];
  const float* ln1w = (const float*)d_in[3];
  const float* ln1b = (const float*)d_in[4];
  const float* Wq   = (const float*)d_in[5];
  const float* bq   = (const float*)d_in[6];
  const float* Wk   = (const float*)d_in[7];
  const float* bk   = (const float*)d_in[8];
  const float* Wv   = (const float*)d_in[9];
  const float* bv   = (const float*)d_in[10];
  const float* Wo   = (const float*)d_in[11];
  const float* bo   = (const float*)d_in[12];
  const float* ln2w = (const float*)d_in[13];
  const float* ln2b = (const float*)d_in[14];
  const float* W1   = (const float*)d_in[15];
  const float* b1   = (const float*)d_in[16];
  const float* W2   = (const float*)d_in[17];
  const float* b2   = (const float*)d_in[18];
  const float* lnfw = (const float*)d_in[19];
  const float* lnfb = (const float*)d_in[20];
  const float* headW= (const float*)d_in[21];
  float* out = (float*)d_out;

  const unsigned T = Bc * Sc;               // 2048 token rows
  const size_t TE = (size_t)T * Ec;
  const size_t SS = (size_t)Bc * Hc * Sc * Sc;
  // fp32 region
  float* x      = (float*)d_ws;             // T*E   (residual stream)
  float* scores = x + TE;                   // B*H*S*S
  // bf16 region
  unsigned short* hb    = (unsigned short*)(scores + SS);
  unsigned short* qb    = hb   + TE;
  unsigned short* kb    = qb   + TE;
  unsigned short* vb    = kb   + TE;
  unsigned short* yb    = vb   + TE;
  unsigned short* ffnb  = yb   + TE;        // T*F
  unsigned short* prb   = ffnb + (size_t)T * Fc;  // B*H*S*S bf16 probs
  unsigned short* bw    = prb + SS;
  const size_t EE = (size_t)Ec * Ec, EF = (size_t)Ec * Fc;
  unsigned short* Wqb = bw;                 // [L][E][E] (N,K)
  unsigned short* Wkb = Wqb + Lc * EE;
  unsigned short* Wvb = Wkb + Lc * EE;
  unsigned short* Wob = Wvb + Lc * EE;
  unsigned short* W1b = Wob + Lc * EE;      // [L][F][E]
  unsigned short* W2b = W1b + Lc * EF;      // [L][E][F]
  unsigned short* headb = W2b + Lc * EF;    // [V][E]

  dim3 blk(256);
  // one-time (per launch) weight conversion to bf16, pre-transposed to [N,K]
  for (int l = 0; l < Lc; ++l) {
    dim3 gT(Ec / 64, Ec / 64);
    conv_t_kernel<<<gT, blk, 0, stream>>>(Wq + l * EE, Wqb + l * EE, Ec, Ec);
    conv_t_kernel<<<gT, blk, 0, stream>>>(Wk + l * EE, Wkb + l * EE, Ec, Ec);
    conv_t_kernel<<<gT, blk, 0, stream>>>(Wv + l * EE, Wvb + l * EE, Ec, Ec);
    conv_t_kernel<<<gT, blk, 0, stream>>>(Wo + l * EE, Wob + l * EE, Ec, Ec);
    conv_t_kernel<<<dim3(Fc / 64, Ec / 64), blk, 0, stream>>>(W1 + l * EF, W1b + l * EF, Ec, Fc);
    conv_t_kernel<<<dim3(Ec / 64, Fc / 64), blk, 0, stream>>>(W2 + l * EF, W2b + l * EF, Fc, Ec);
  }
  {
    long long n4 = ((long long)Vc * Ec) / 4;
    conv_kernel<<<dim3((unsigned)((n4 + 255) / 256)), blk, 0, stream>>>(headW, headb, n4);
  }

  embed_kernel<<<dim3(T), blk, 0, stream>>>(idx, tok, pos, x);

  dim3 gE(Ec / 128, T / 128);
  for (int l = 0; l < Lc; ++l) {
    const size_t e = (size_t)l * Ec, f = (size_t)l * Fc;

    ln_kernel<<<dim3(T), blk, 0, stream>>>(x, ln1w + e, ln1b + e, hb);
    gemm_wmma<<<gE, blk, 0, stream>>>(hb, Wqb + l * EE, bq + e, nullptr, qb, (int)T, Ec, Ec, 0, 1);
    gemm_wmma<<<gE, blk, 0, stream>>>(hb, Wkb + l * EE, bk + e, nullptr, kb, (int)T, Ec, Ec, 0, 1);
    gemm_wmma<<<gE, blk, 0, stream>>>(hb, Wvb + l * EE, bv + e, nullptr, vb, (int)T, Ec, Ec, 0, 1);

    attn_scores<<<dim3(Sc / 128, Sc / 128, Bc * Hc), blk, 0, stream>>>(qb, kb, scores, 0.125f);
    softmax_kernel<<<dim3(Sc, Bc * Hc), blk, 0, stream>>>(scores, prb);
    attn_av<<<dim3(Sc / 128, Bc * Hc), blk, 0, stream>>>(prb, vb, yb);

    gemm_wmma<<<gE, blk, 0, stream>>>(yb, Wob + l * EE, bo + e, x, x, (int)T, Ec, Ec, 0, 0);
    ln_kernel<<<dim3(T), blk, 0, stream>>>(x, ln2w + e, ln2b + e, hb);
    gemm_wmma<<<dim3(Fc / 128, T / 128), blk, 0, stream>>>(hb, W1b + l * EF, b1 + f, nullptr, ffnb, (int)T, Fc, Ec, 1, 1);
    gemm_wmma<<<gE, blk, 0, stream>>>(ffnb, W2b + l * EF, b2 + e, x, x, (int)T, Ec, Fc, 0, 0);
  }

  ln_kernel<<<dim3(T), blk, 0, stream>>>(x, lnfw, lnfb, hb);
  gemm_wmma<<<dim3((Vc + 127) / 128, T / 128), blk, 0, stream>>>(
      hb, headb, nullptr, nullptr, out, (int)T, Vc, Ec, 0, 0);
}
